// Attention_78512002171348
// MI455X (gfx1250) — matrix operand order repaired and checked by reading, not compiled, not run on hardware
//
#include <hip/hip_runtime.h>
#include <hip/hip_bf16.h>

// ---------------------------------------------------------------------------
// Types for CDNA5 WMMA (wave32): V_WMMA_F32_16X16X32_BF16
// ---------------------------------------------------------------------------
typedef __bf16 v16bf __attribute__((ext_vector_type(16)));
typedef __bf16 v8bf  __attribute__((ext_vector_type(8)));
typedef float  v8f   __attribute__((ext_vector_type(8)));
typedef int    v4i   __attribute__((ext_vector_type(4)));

#define WMMA_BF16(a, b, c) \
  __builtin_amdgcn_wmma_f32_16x16x32_bf16(false, (a), false, (b), (short)0, (c), false, false)

// Problem constants (match reference)
#define BB   4
#define SS   2048
#define DD   1024
#define HH   16
#define DH   64
#define MM   (BB * SS)     // 8192 rows

// ---------------------------------------------------------------------------
// CDNA5 async global->LDS (ASYNCcnt) with synchronous fallback
// ---------------------------------------------------------------------------
#define AS1 __attribute__((address_space(1)))
#define AS3 __attribute__((address_space(3)))

#if defined(__gfx1250__) && __has_builtin(__builtin_amdgcn_global_load_async_to_lds_b128)
#define HAVE_ASYNC_LDS 1
#endif

static __device__ __forceinline__ void copy16_g2l(const __bf16* g, __bf16* l) {
#if defined(HAVE_ASYNC_LDS)
  __builtin_amdgcn_global_load_async_to_lds_b128((AS1 v4i*)(const void*)g,
                                                 (AS3 v4i*)(void*)l, 0, 0);
#else
  *(uint4*)l = *(const uint4*)g;
#endif
}

static __device__ __forceinline__ void async_wait_all() {
#if defined(HAVE_ASYNC_LDS)
#if __has_builtin(__builtin_amdgcn_s_wait_asynccnt)
  __builtin_amdgcn_s_wait_asynccnt(0);
#else
  asm volatile("s_wait_asynccnt 0x0" ::: "memory");
#endif
#endif
}

// A-fragment (16x32 bf16, MxK): lane holds row M=lane%16; elems 0-7 = K[8*half..],
// elems 8-15 = K[16+8*half..]  (half = lane>>4). Two 16B loads.
static __device__ __forceinline__ v16bf load_a_frag(const __bf16* row, int half) {
  v16bf a;
  *(v8bf*)&a       = *(const v8bf*)(row + half * 8);
  *((v8bf*)&a + 1) = *(const v8bf*)(row + 16 + half * 8);
  return a;
}

// B-fragment (32x16 bf16, KxN, column-major): lane holds column N=lane%16;
// elems 0-7 = K[8*half..], elems 8-15 = K[16+8*half..]. Two 16B loads (padded-stride safe).
static __device__ __forceinline__ v16bf load_b_frag(const __bf16* col, int half) {
  v16bf b;
  *(v8bf*)&b       = *(const v8bf*)(col + half * 8);
  *((v8bf*)&b + 1) = *(const v8bf*)(col + 16 + half * 8);
  return b;
}

// ---------------------------------------------------------------------------
// Kernel 0a: fp32 -> bf16 (contiguous), 4 elems/thread
// ---------------------------------------------------------------------------
__global__ __launch_bounds__(256) void cvt_bf16_kernel(const float* __restrict__ in,
                                                       __bf16* __restrict__ out) {
  size_t i = ((size_t)blockIdx.x * 256 + threadIdx.x) * 4;
  float4 f = *(const float4*)(in + i);
  out[i + 0] = (__bf16)f.x;
  out[i + 1] = (__bf16)f.y;
  out[i + 2] = (__bf16)f.z;
  out[i + 3] = (__bf16)f.w;
}

// ---------------------------------------------------------------------------
// Kernel 0b: fp32 [K][N] -> bf16 [N][K] (transpose via LDS, 32x32 tiles)
// ---------------------------------------------------------------------------
__global__ __launch_bounds__(256) void cvt_transpose_kernel(const float* __restrict__ in,
                                                            __bf16* __restrict__ out,
                                                            int K, int N) {
  __shared__ float tile[32][33];
  const int k0 = blockIdx.y * 32, n0 = blockIdx.x * 32;
  const int tx = threadIdx.x & 31, ty = threadIdx.x >> 5;  // ty in [0,8)
#pragma unroll
  for (int i = 0; i < 4; ++i)
    tile[ty + 8 * i][tx] = in[(size_t)(k0 + ty + 8 * i) * N + n0 + tx];
  __syncthreads();
#pragma unroll
  for (int i = 0; i < 4; ++i)
    out[(size_t)(n0 + ty + 8 * i) * K + k0 + tx] = (__bf16)tile[tx][ty + 8 * i];
}

// ---------------------------------------------------------------------------
// Kernel 1/3: WMMA GEMM.  C[M,N] = A[M,K] * Bt[N,K]^T
//   mode 0: QKV projection -> scatter q[b,h,s,d], k[b,h,s,d], vt[b,h,d,s] (bf16)
//   mode 1: out projection -> fout[m,n] = acc + bias[n]                  (f32)
// Tile: BM=128 x BN=128, BK=32 (one WMMA K-step). 8 waves; each wave owns a
// 32x64 patch = 2x4 16x16 subtiles (8 WMMAs / 12 fragment loads per K-step).
// LDS rows padded to 40 elems (80B) -> fragment row-reads are bank-conflict-free.
// Async global->LDS double buffering, 1 barrier per K-step.
// ---------------------------------------------------------------------------
#define BM 128
#define BN 128
#define BK 32
#define LDK 40   // padded LDS row stride (elems): 80B, odd multiple of 16B

__global__ __launch_bounds__(256) void gemm_wmma_bf16_kernel(
    int mode,
    const __bf16* __restrict__ A, const __bf16* __restrict__ Bt,
    const float* __restrict__ bias,
    __bf16* __restrict__ qout, __bf16* __restrict__ kout, __bf16* __restrict__ vtout,
    float* __restrict__ fout,
    int N, int K) {
  __shared__ __bf16 As[2][BM][LDK];
  __shared__ __bf16 Bs[2][BN][LDK];

  const int tid  = threadIdx.x;
  const int lane = tid & 31;
  const int ln   = lane & 15;
  const int half = lane >> 4;
  const int wave = tid >> 5;   // 0..7
  const int wm   = wave >> 1;  // 0..3  (32-row strip)
  const int wn   = wave & 1;   // 0..1  (64-col strip)

  const int m0 = blockIdx.y * BM;
  const int n0 = blockIdx.x * BN;
  const int nk = K / BK;

  v8f acc[2][4];
  const v8f vzero = {0.f, 0.f, 0.f, 0.f, 0.f, 0.f, 0.f, 0.f};
#pragma unroll
  for (int mt = 0; mt < 2; ++mt)
#pragma unroll
    for (int nt = 0; nt < 4; ++nt) acc[mt][nt] = vzero;

  // staging map: A tile = 128x32 = 512 x 16B chunks, 2/thread; B tile identical
  const int r0 = tid >> 2, r1 = (tid + 256) >> 2;
  const int cc = (tid & 3) * 8;

  auto stage = [&](int buf, int kt) {
    const size_t ko = (size_t)kt * BK + cc;
    copy16_g2l(A  + (size_t)(m0 + r0) * K + ko, &As[buf][r0][cc]);
    copy16_g2l(A  + (size_t)(m0 + r1) * K + ko, &As[buf][r1][cc]);
    copy16_g2l(Bt + (size_t)(n0 + r0) * K + ko, &Bs[buf][r0][cc]);
    copy16_g2l(Bt + (size_t)(n0 + r1) * K + ko, &Bs[buf][r1][cc]);
  };

  stage(0, 0);
  int buf = 0;
  for (int kt = 0; kt < nk; ++kt) {
    async_wait_all();     // my async loads into `buf` done
    __syncthreads();      // everyone's loads done; everyone done reading buf^1
    if (kt + 1 < nk) {
      stage(buf ^ 1, kt + 1);
      if (kt + 2 < nk)    // pull next-next lines toward GL2 (global_prefetch_b8)
        __builtin_prefetch(A + (size_t)(m0 + r0) * K + (size_t)(kt + 2) * BK + cc, 0, 0);
    }

    v16bf afrag[2];
#pragma unroll
    for (int mt = 0; mt < 2; ++mt)
      afrag[mt] = load_a_frag(&As[buf][wm * 32 + mt * 16 + ln][0], half);
#pragma unroll
    for (int nt = 0; nt < 4; ++nt) {
      v16bf bfrag = load_b_frag(&Bs[buf][wn * 64 + nt * 16 + ln][0], half);
#pragma unroll
      for (int mt = 0; mt < 2; ++mt)
        acc[mt][nt] = WMMA_BF16(afrag[mt], bfrag, acc[mt][nt]);
    }
    buf ^= 1;
  }

  // ---- epilogue (C layout: VGPR r, lane -> M = r + 8*half, N = ln) ----
  const int mb = m0 + wm * 32;
  const int nb = n0 + wn * 64;
#pragma unroll
  for (int mt = 0; mt < 2; ++mt) {
#pragma unroll
    for (int nt = 0; nt < 4; ++nt) {
      v8f c = acc[mt][nt];
      const int ncol  = nb + nt * 16 + ln;
      const int mrow0 = mb + mt * 16 + 8 * half;
      if (mode == 0) {
        const int sel = ncol >> 10;        // 0=q, 1=k, 2=v (uniform per subtile)
        const int nn  = ncol & 1023;
        const int h = nn >> 6, d = nn & 63;
#pragma unroll
        for (int r = 0; r < 8; ++r) {
          const int m = mrow0 + r;
          const int b = m >> 11, s = m & 2047;
          const __bf16 val = (__bf16)c[r];
          if (sel == 0)      qout [((size_t)((b << 4) + h) * SS + s) * DH + d] = val;
          else if (sel == 1) kout [((size_t)((b << 4) + h) * SS + s) * DH + d] = val;
          else               vtout[((size_t)((b << 4) + h) * DH + d) * SS + s] = val;
        }
      } else {
        const float bv = bias[ncol];
#pragma unroll
        for (int r = 0; r < 8; ++r)
          fout[(size_t)(mrow0 + r) * N + ncol] = c[r] + bv;
      }
    }
  }
}

// ---------------------------------------------------------------------------
// Kernel 2: flash attention.  grid = (S/64, H, B), 4 waves (128 thr).
// Q A-frags in registers for the whole kernel; 32-key K / V^T blocks staged
// via async global->LDS double buffering; online softmax with __shfl_xor row
// reductions; P routed through per-wave LDS scratch (C-layout -> A-layout).
// ---------------------------------------------------------------------------
#define KLD 72   // Ks padded stride (elems): 144B
#define VLD 40   // Vs/Ps padded stride (elems): 80B

__global__ __launch_bounds__(128) void flash_attn_kernel(
    const __bf16* __restrict__ q,   // [b,h,s,d]
    const __bf16* __restrict__ k,   // [b,h,s,d]
    const __bf16* __restrict__ vt,  // [b,h,d,s]
    __bf16* __restrict__ ctx)       // [b*s, h*64+d]
{
  const int lane = threadIdx.x & 31;
  const int ln   = lane & 15;
  const int half = lane >> 4;
  const int wave = threadIdx.x >> 5;  // 0..3
  const int b = blockIdx.z, h = blockIdx.y;
  const int bh = b * HH + h;
  const int s0 = blockIdx.x * 64;

  const __bf16* qbase = q  + (size_t)bh * SS * DH;
  const __bf16* kbase = k  + (size_t)bh * SS * DH;
  const __bf16* vbase = vt + (size_t)bh * DH * SS;

  __shared__ __bf16 Ks[2][32][KLD];   // 32 keys x 64 d   (B-frag for Q@K^T)
  __shared__ __bf16 Vs[2][DH][VLD];   // 64 d   x 32 keys (B-frag for P@V)
  __shared__ __bf16 Ps[4][16][VLD];   // per-wave P scratch

  // Q A-fragments for this wave's 16 rows, d-chunks {0..31},{32..63}
  v16bf aq[2];
  {
    const __bf16* qr = qbase + (size_t)(s0 + wave * 16 + ln) * DH;
#pragma unroll
    for (int c = 0; c < 2; ++c) aq[c] = load_a_frag(qr + c * 32, half);
  }

  const v8f vzero = {0.f, 0.f, 0.f, 0.f, 0.f, 0.f, 0.f, 0.f};
  v8f o[4];
#pragma unroll
  for (int dt = 0; dt < 4; ++dt) o[dt] = vzero;
  float rm[8], rl[8];
#pragma unroll
  for (int r = 0; r < 8; ++r) { rm[r] = -3.0e38f; rl[r] = 0.f; }

  auto stageKV = [&](int buf, int kb) {
#pragma unroll
    for (int i = 0; i < 2; ++i) {
      const int c = threadIdx.x + i * 128;
      const int kr = c >> 3, kc = (c & 7) * 8;
      copy16_g2l(kbase + (size_t)(kb + kr) * DH + kc, &Ks[buf][kr][kc]);
      const int vr = c >> 2, vc = (c & 3) * 8;
      copy16_g2l(vbase + (size_t)vr * SS + kb + vc, &Vs[buf][vr][vc]);
    }
  };

  stageKV(0, 0);
  int buf = 0;
  for (int kb = 0; kb < SS; kb += 32) {
    async_wait_all();
    __syncthreads();
    if (kb + 32 < SS) stageKV(buf ^ 1, kb + 32);

    // scores: 16 q x 32 keys = two 16x16 WMMA tiles, K-dim = 64 (2 steps)
    v8f sc[2];
#pragma unroll
    for (int nt = 0; nt < 2; ++nt) {
      v8f a = vzero;
#pragma unroll
      for (int c = 0; c < 2; ++c) {
        v16bf bk = load_b_frag(&Ks[buf][nt * 16 + ln][c * 32], half);
        a = WMMA_BF16(aq[c], bk, a);
      }
      sc[nt] = a;
    }

    // online softmax per row (row M = r + 8*half lives in one 16-lane group)
#pragma unroll
    for (int r = 0; r < 8; ++r) {
      float s0v = sc[0][r] * 0.125f;   // 1/sqrt(64)
      float s1v = sc[1][r] * 0.125f;
      float bmx = fmaxf(s0v, s1v);
#pragma unroll
      for (int m = 8; m >= 1; m >>= 1) bmx = fmaxf(bmx, __shfl_xor(bmx, m, 32));
      const float mnew = fmaxf(rm[r], bmx);
      const float corr = __expf(rm[r] - mnew);
      const float p0 = __expf(s0v - mnew);
      const float p1 = __expf(s1v - mnew);
      float ps = p0 + p1;
#pragma unroll
      for (int m = 8; m >= 1; m >>= 1) ps += __shfl_xor(ps, m, 32);
      rl[r] = rl[r] * corr + ps;
      rm[r] = mnew;
#pragma unroll
      for (int dt = 0; dt < 4; ++dt) o[dt][r] *= corr;
      Ps[wave][8 * half + r][ln]      = (__bf16)p0;
      Ps[wave][8 * half + r][16 + ln] = (__bf16)p1;
    }

    // C-layout -> A-layout via per-wave LDS (in-wave DS ordering, no barrier)
    v16bf pa = load_a_frag(&Ps[wave][ln][0], half);

    // O += P @ V  (four 16x16 d-tiles, K-dim = 32 keys)
#pragma unroll
    for (int dt = 0; dt < 4; ++dt) {
      v16bf bv = load_b_frag(&Vs[buf][dt * 16 + ln][0], half);
      o[dt] = WMMA_BF16(pa, bv, o[dt]);
    }
    buf ^= 1;
  }

  // normalize + store ctx as [b*S + s][h*64 + d] bf16
  float invl[8];
#pragma unroll
  for (int r = 0; r < 8; ++r) invl[r] = 1.0f / rl[r];
  const int srow0 = s0 + wave * 16 + 8 * half;
#pragma unroll
  for (int dt = 0; dt < 4; ++dt) {
#pragma unroll
    for (int r = 0; r < 8; ++r) {
      const int s = srow0 + r;
      ctx[(size_t)(b * SS + s) * DD + h * DH + dt * 16 + ln] = (__bf16)(o[dt][r] * invl[r]);
    }
  }
}

// ---------------------------------------------------------------------------
// Launch: cvt -> QKV GEMM -> flash attention -> out GEMM (+bias)
// ---------------------------------------------------------------------------
extern "C" void kernel_launch(void* const* d_in, const int* in_sizes, int n_in,
                              void* d_out, int out_size, void* d_ws, size_t ws_size,
                              hipStream_t stream) {
  const float* x     = (const float*)d_in[0];   // [4,2048,1024]
  const float* w_q   = (const float*)d_in[1];   // [1024,1024]  (K,N)
  const float* w_kv  = (const float*)d_in[2];   // [1024,2048]  (K,N)
  const float* w_out = (const float*)d_in[3];   // [1024,1024]  (K,N)
  const float* b_out = (const float*)d_in[4];   // [1024]
  float* out = (float*)d_out;                   // [4,2048,1024]

  char* w = (char*)d_ws;
  __bf16* xb     = (__bf16*)(w);                       w += (size_t)MM * DD * 2;       // 16 MB
  __bf16* wqkvT  = (__bf16*)(w);                       w += (size_t)3072 * DD * 2;     //  6 MB
  __bf16* wotT   = (__bf16*)(w);                       w += (size_t)DD * DD * 2;       //  2 MB
  __bf16* qb     = (__bf16*)(w);                       w += (size_t)MM * DD * 2;
  __bf16* kbuf   = (__bf16*)(w);                       w += (size_t)MM * DD * 2;
  __bf16* vtb    = (__bf16*)(w);                       w += (size_t)MM * DD * 2;
  __bf16* ctx    = (__bf16*)(w);                       w += (size_t)MM * DD * 2;

  // 0) precision conversion (x contiguous; weights transposed to [N][K])
  cvt_bf16_kernel<<<(MM * DD) / 1024, 256, 0, stream>>>(x, xb);
  cvt_transpose_kernel<<<dim3(1024 / 32, 1024 / 32), 256, 0, stream>>>(w_q,   wqkvT,               1024, 1024);
  cvt_transpose_kernel<<<dim3(2048 / 32, 1024 / 32), 256, 0, stream>>>(w_kv,  wqkvT + 1024 * 1024, 1024, 2048);
  cvt_transpose_kernel<<<dim3(1024 / 32, 1024 / 32), 256, 0, stream>>>(w_out, wotT,                1024, 1024);

  // 1) fused QKV projection: [8192,1024] x [1024,3072] -> q/k/vt (bf16)
  gemm_wmma_bf16_kernel<<<dim3(3072 / BN, MM / BM), 256, 0, stream>>>(
      0, xb, wqkvT, nullptr, qb, kbuf, vtb, nullptr, 3072, DD);

  // 2) attention (flash-style, per head, 64-row q tiles)
  flash_attn_kernel<<<dim3(SS / 64, HH, BB), 128, 0, stream>>>(qb, kbuf, vtb, ctx);

  // 3) output projection + bias, fp32 out: [8192,1024] x [1024,1024]
  gemm_wmma_bf16_kernel<<<dim3(DD / BN, MM / BM), 256, 0, stream>>>(
      1, ctx, wotT, b_out, nullptr, nullptr, nullptr, out, DD, DD);

  (void)in_sizes; (void)n_in; (void)out_size; (void)ws_size;
}
